// LinearAttention_31928786878890
// MI455X (gfx1250) — matrix operand order repaired and checked by reading, not compiled, not run on hardware
//
#include <hip/hip_runtime.h>
#include <hip/hip_bf16.h>

// ---------------------------------------------------------------------------
// Problem constants (from reference)
// ---------------------------------------------------------------------------
#define BATCH   8
#define CIN     512
#define ZLEN    4096
#define HEADS   8
#define DHEAD   64
#define O3      1536          // 3 * HEADS * DHEAD
#define HID     512           // HEADS * DHEAD
#define QSCALE  0.125f        // DHEAD^-0.5

// ---------------------------------------------------------------------------
// WMMA types / helpers (CDNA5 wave32, V_WMMA_F32_16X16X32_BF16)
// ---------------------------------------------------------------------------
typedef __attribute__((ext_vector_type(16))) __bf16 v16bf;
typedef __attribute__((ext_vector_type(8)))  __bf16 bf16x8;
typedef __attribute__((ext_vector_type(8)))  float  v8f;
typedef __attribute__((ext_vector_type(4)))  float  f32x4;

__device__ __forceinline__ v8f wmma_bf16(v16bf a, v16bf b, v8f c) {
  // (neg_a, A, neg_b, B, c_mod, C, reuse_a, reuse_b)
  return __builtin_amdgcn_wmma_f32_16x16x32_bf16(false, a, false, b, (short)0, c,
                                                 false, false);
}

__device__ __forceinline__ v16bf combine8(bf16x8 lo, bf16x8 hi) {
  v16bf f;
#pragma unroll
  for (int i = 0; i < 8; ++i) { f[i] = lo[i]; f[i + 8] = hi[i]; }
  return f;
}

// A fragment (16x32 bf16, M x K): lane&15 = row M; lanes 0-15 hold
// K[kb..kb+15] contiguous in v0-7 with kb = (lane>>4)*16.
// LD must give 16B-aligned rows so the two 8-half chunks lower to b128 reads.
__device__ __forceinline__ v16bf fragA_lds(const __bf16* t, int ld, int lane) {
  const int r  = lane & 15;
  const int kb = (lane >> 4) << 4;
  const __bf16* p = t + r * ld + kb;
  v16bf f;
#pragma unroll
  for (int i = 0; i < 8; ++i) {
    f[i]     = p[i];
    f[i + 8] = p[8 + i];
  }
  return f;
}

// B fragment (32x16 bf16, K x N) from a TRANSPOSED [n][k] LDS tile:
// lane&15 = column N; lanes 0-15 hold K=0..15, lanes 16-31 hold K=16..31
// => 16 contiguous halves (two b128 LDS reads) per lane.
__device__ __forceinline__ v16bf fragB_ldsT(const __bf16* t, int ld, int lane) {
  const int c  = lane & 15;
  const int kb = (lane >> 4) << 4;
  const __bf16* p = t + c * ld + kb;
  v16bf f;
#pragma unroll
  for (int i = 0; i < 16; ++i) f[i] = p[i];
  return f;
}

// ---------------------------------------------------------------------------
// CDNA5 async global->LDS copy (16 bytes / lane) + ASYNCcnt wait.
// VGLOBAL async encoding: VDST = LDS byte address VGPR, VADDR = 64-bit addr.
// ---------------------------------------------------------------------------
__device__ __forceinline__ void async_copy_b128(void* lds_dst,
                                                const void* gsrc) {
  const unsigned int       l = (unsigned int)(unsigned long long)lds_dst;
  const unsigned long long g = (unsigned long long)gsrc;
  asm volatile("global_load_async_to_lds_b128 %0, %1, off"
               :
               : "v"(l), "v"(g)
               : "memory");
}

__device__ __forceinline__ void wait_async0() {
  asm volatile("s_wait_asynccnt 0x0" ::: "memory");
}

// ---------------------------------------------------------------------------
// Kernel 1/4: generic per-batch GEMM  C[bz] = A(MxK) * B[bz](KxN)
//   A: f32 row-major (shared over batch), bf16-converted while staging.
//   B: f32 or bf16 row-major per batch, staged TRANSPOSED into LDS.
//   C: bf16 or f32 per batch; optional bias[M]; optional QSCALE fold on
//      A-rows < 512 (the q slice of Wqkv).
// Tile: 64(M) x 128(N) per 256-thread block (8 waves in 2x4), BK = 32,
// double-buffered LDS (one barrier per K-step), 4 WMMAs / wave / K-step.
// ---------------------------------------------------------------------------
template <typename TB, bool OUT_BF16, bool ADD_BIAS, bool FOLD_SCALE>
__global__ __launch_bounds__(256) void gemm_bf16_kernel(
    const float* __restrict__ A, const TB* __restrict__ Bmat,
    void* __restrict__ Cmat, const float* __restrict__ bias, int M, int N,
    int K) {
  constexpr int LDA = 40;  // 80B rows (16B multiple)
  constexpr int LDB = 40;  // transposed B tile rows, 80B
  __shared__ __align__(16) __bf16 lds_a[2][64 * LDA];
  __shared__ __align__(16) __bf16 lds_bt[2][128 * LDB];

  const int tid  = threadIdx.x;
  const int lane = tid & 31;
  const int w    = tid >> 5;
  const int wm   = (w & 1) * 32;   // wave's 32 M rows (2 tiles)
  const int wn   = (w >> 1) * 32;  // wave's 32 N cols (2 tiles)
  const int m0   = blockIdx.y * 64;
  const int n0   = blockIdx.x * 128;

  const TB* __restrict__ Bb = Bmat + (size_t)blockIdx.z * K * N;
  const size_t cbase = (size_t)blockIdx.z * M * N;

  // staging assignments
  const int a_row = tid >> 2;            // 64 rows, 4 threads/row
  const int a_col = (tid & 3) << 3;      // 8 halves each
  const int b_n   = tid & 127;           // 128 columns
  const int b_c   = (tid >> 7) << 4;     // k-chunk 0 or 16

  const float a_scale =
      (FOLD_SCALE && (m0 + a_row) < HID) ? QSCALE : 1.0f;

  v8f acc[2][2] = {{{}, {}}, {{}, {}}};

  auto stage = [&](int buf, int k0) {
    // ---- A tile 64x32 f32 -> bf16, row-major --------------------------
    {
      const float* src = A + (size_t)(m0 + a_row) * K + k0 + a_col;
      f32x4 p0 = *(const f32x4*)src;
      f32x4 p1 = *(const f32x4*)(src + 4);
      bf16x8 vv;
#pragma unroll
      for (int i = 0; i < 4; ++i) {
        vv[i]     = (__bf16)(p0[i] * a_scale);
        vv[i + 4] = (__bf16)(p1[i] * a_scale);
      }
      *(bf16x8*)(&lds_a[buf][a_row * LDA + a_col]) = vv;
      if (k0 + 64 < K) __builtin_prefetch(src + 64, 0, 0);
    }
    // ---- B tile 32x128 -> transposed [n][k] ---------------------------
    {
      const TB* src = Bb + (size_t)(k0 + b_c) * N + n0 + b_n;
      bf16x8 v0, v1;
#pragma unroll
      for (int j = 0; j < 8; ++j) v0[j] = (__bf16)(float)src[(size_t)j * N];
#pragma unroll
      for (int j = 0; j < 8; ++j)
        v1[j] = (__bf16)(float)src[(size_t)(j + 8) * N];
      __bf16* d = &lds_bt[buf][b_n * LDB + b_c];
      *(bf16x8*)d       = v0;
      *(bf16x8*)(d + 8) = v1;
    }
  };

  stage(0, 0);
  __syncthreads();

  int buf = 0;
  for (int k0 = 0; k0 < K; k0 += 32) {
    if (k0 + 32 < K) stage(buf ^ 1, k0 + 32);

    const v16bf a0 = fragA_lds(&lds_a[buf][(wm + 0) * LDA], LDA, lane);
    const v16bf a1 = fragA_lds(&lds_a[buf][(wm + 16) * LDA], LDA, lane);
    const v16bf b0 = fragB_ldsT(&lds_bt[buf][(wn + 0) * LDB], LDB, lane);
    const v16bf b1 = fragB_ldsT(&lds_bt[buf][(wn + 16) * LDB], LDB, lane);
    acc[0][0] = wmma_bf16(a0, b0, acc[0][0]);
    acc[0][1] = wmma_bf16(a0, b1, acc[0][1]);
    acc[1][0] = wmma_bf16(a1, b0, acc[1][0]);
    acc[1][1] = wmma_bf16(a1, b1, acc[1][1]);

    __syncthreads();
    buf ^= 1;
  }

  // ---- epilogue: D layout => vgpr r: M = r + (lane>=16)*8, N = lane&15
  const int hi8 = (lane >> 4) << 3;
  const int cN  = lane & 15;
#pragma unroll
  for (int mi = 0; mi < 2; ++mi) {
#pragma unroll
    for (int ni = 0; ni < 2; ++ni) {
      const int row0 = m0 + wm + mi * 16 + hi8;
      const int col  = n0 + wn + ni * 16 + cN;
#pragma unroll
      for (int r = 0; r < 8; ++r) {
        const int row = row0 + r;
        float v = acc[mi][ni][r];
        if constexpr (ADD_BIAS) v += bias[row];
        if constexpr (OUT_BF16) {
          ((__bf16*)Cmat + cbase)[(size_t)row * N + col] = (__bf16)v;
        } else {
          ((float*)Cmat + cbase)[(size_t)row * N + col] = v;
        }
      }
    }
  }
}

// ---------------------------------------------------------------------------
// Kernel 2: in-place softmax over z for the k slice of qkv (bf16, f32 math).
// One 256-thread block per row (BATCH*HID = 4096 rows of length ZLEN).
// ---------------------------------------------------------------------------
__global__ __launch_bounds__(256) void softmax_k_kernel(
    __bf16* __restrict__ qkv) {
  const int blk = blockIdx.x;
  const int b   = blk >> 9;  // / 512
  const int r   = blk & 511;
  __bf16* row = qkv + ((size_t)b * O3 + HID + r) * ZLEN;

  __shared__ float red[256];
  const int t = threadIdx.x;

  float vals[16];
  float mx = -3.0e38f;
#pragma unroll
  for (int i = 0; i < 16; ++i) {
    vals[i] = (float)row[t + 256 * i];
    mx = fmaxf(mx, vals[i]);
  }
  red[t] = mx;
  __syncthreads();
  for (int s = 128; s > 0; s >>= 1) {
    if (t < s) red[t] = fmaxf(red[t], red[t + s]);
    __syncthreads();
  }
  mx = red[0];
  __syncthreads();

  float sum = 0.0f;
#pragma unroll
  for (int i = 0; i < 16; ++i) {
    vals[i] = __expf(vals[i] - mx);
    sum += vals[i];
  }
  red[t] = sum;
  __syncthreads();
  for (int s = 128; s > 0; s >>= 1) {
    if (t < s) red[t] += red[t + s];
    __syncthreads();
  }
  const float inv = 1.0f / red[0];
#pragma unroll
  for (int i = 0; i < 16; ++i) row[t + 256 * i] = (__bf16)(vals[i] * inv);
}

// ---------------------------------------------------------------------------
// Kernel 3: per-(b,h) fused linear attention.
//   phase 1: ctx[d][e] = sum_n k_sm[d,n] * v[e,n]   (both K-contiguous in
//            global -> WMMA fragments straight from global b128 loads)
//   phase 2: out[e][n] = sum_d ctx[d][e] * q[d,n]   (ctx in LDS; q tile
//            staged via GLOBAL_LOAD_ASYNC_TO_LDS_B128 + s_wait_asynccnt)
// Grid: 64 blocks (b*8+h) x 256 threads (8 waves).
// ---------------------------------------------------------------------------
__global__ __launch_bounds__(256) void attention_kernel(
    const __bf16* __restrict__ qkv, __bf16* __restrict__ att) {
  const int bh = blockIdx.x;
  const int b  = bh >> 3;
  const int h  = bh & 7;
  const size_t base = (size_t)b * O3 * ZLEN;
  const __bf16* qp = qkv + base + (size_t)(h * DHEAD) * ZLEN;
  const __bf16* kp = qkv + base + (size_t)(HID + h * DHEAD) * ZLEN;
  const __bf16* vp = qkv + base + (size_t)(2 * HID + h * DHEAD) * ZLEN;
  __bf16* op = att + ((size_t)b * HID + h * DHEAD) * ZLEN;

  constexpr int LDC = 68;   // ctx f32 64x64, padded
  constexpr int LDQ = 136;  // q tile bf16 64x128, padded (272B = 16B mult.)
  __shared__ __align__(16) float  ctx[64 * LDC];
  __shared__ __align__(16) __bf16 qt[64 * LDQ];

  const int tid  = threadIdx.x;
  const int lane = tid & 31;
  const int w    = tid >> 5;

  // ---------------- phase 1: ctx = k · v^T (M=d, N=e, K=n) ----------------
  {
    const int dt = (w & 3) * 16;        // d tile
    const int et = (w >> 2) * 32;       // e tiles (2 x 16)
    const int r    = lane & 15;
    const int kb8  = (lane >> 4) << 3;  // unused; A also loads 16 contiguous K at kb16
    const int kb16 = (lane >> 4) << 4;  // B: 16 contiguous K
    const __bf16* ka  = kp + (size_t)(dt + r) * ZLEN;
    const __bf16* va0 = vp + (size_t)(et + r) * ZLEN;
    const __bf16* va1 = vp + (size_t)(et + 16 + r) * ZLEN;

    v8f c0 = {};
    v8f c1 = {};
    for (int n0 = 0; n0 < ZLEN; n0 += 32) {
      const v16bf af = combine8(*(const bf16x8*)(ka + n0 + kb16),
                                *(const bf16x8*)(ka + n0 + kb16 + 8));
      const v16bf b0 = combine8(*(const bf16x8*)(va0 + n0 + kb16),
                                *(const bf16x8*)(va0 + n0 + kb16 + 8));
      const v16bf b1 = combine8(*(const bf16x8*)(va1 + n0 + kb16),
                                *(const bf16x8*)(va1 + n0 + kb16 + 8));
      c0 = wmma_bf16(af, b0, c0);
      c1 = wmma_bf16(af, b1, c1);
    }
    const int d0 = dt + ((lane >> 4) << 3);
    const int e0 = et + (lane & 15);
#pragma unroll
    for (int i = 0; i < 8; ++i) {
      ctx[(d0 + i) * LDC + e0]      = c0[i];
      ctx[(d0 + i) * LDC + e0 + 16] = c1[i];
    }
  }
  __syncthreads();

  // ---------------- phase 2: out = ctx^T · q (M=e, N=n, K=d) --------------
  // Hoist the 4 e-tiles x 2 k-halves of the ctx^T A-fragments.
  v16bf afr[4][2];
  {
    const int ecol = lane & 15;
    const int kb8  = (lane >> 4) << 4;
#pragma unroll
    for (int e2 = 0; e2 < 4; ++e2) {
#pragma unroll
      for (int kd = 0; kd < 2; ++kd) {
        const int kbase = kd * 32 + kb8;
        v16bf f;
#pragma unroll
        for (int i = 0; i < 8; ++i) {
          f[i]     = (__bf16)ctx[(kbase + i) * LDC + e2 * 16 + ecol];
          f[i + 8] = (__bf16)ctx[(kbase + 8 + i) * LDC + e2 * 16 + ecol];
        }
        afr[e2][kd] = f;
      }
    }
  }

  const int ns = w * 16;  // this wave's n columns inside the 128-wide block
  for (int n0 = 0; n0 < ZLEN; n0 += 128) {
    __syncthreads();  // previous iteration's qt readers are done
    // stage q block [64 d][128 n] into LDS with async b128 DMA
#pragma unroll
    for (int idx = tid; idx < 64 * 16; idx += 256) {
      const int row = idx >> 4;
      const int col = (idx & 15) << 3;
      async_copy_b128(qt + row * LDQ + col,
                      qp + (size_t)row * ZLEN + n0 + col);
    }
    wait_async0();
    __syncthreads();

#pragma unroll
    for (int e2 = 0; e2 < 4; ++e2) {
      v8f acc = {};
#pragma unroll
      for (int kd = 0; kd < 2; ++kd) {
        const int cc   = ns + (lane & 15);
        const int kb16 = kd * 32 + ((lane >> 4) << 4);
        v16bf bfr;
#pragma unroll
        for (int i = 0; i < 16; ++i) bfr[i] = qt[(kb16 + i) * LDQ + cc];
        acc = wmma_bf16(afr[e2][kd], bfr, acc);
      }
      const int e0 = e2 * 16 + ((lane >> 4) << 3);
      const int cc = n0 + ns + (lane & 15);
#pragma unroll
      for (int i = 0; i < 8; ++i)
        op[(size_t)(e0 + i) * ZLEN + cc] = (__bf16)acc[i];
    }
  }
}

// ---------------------------------------------------------------------------
// Launch: qkv GEMM -> softmax(k) -> fused attention -> output GEMM (+bias)
// Workspace: qkv bf16 [B,1536,Z] (96 MiB) + att bf16 [B,512,Z] (32 MiB)
// ---------------------------------------------------------------------------
extern "C" void kernel_launch(void* const* d_in, const int* in_sizes, int n_in,
                              void* d_out, int out_size, void* d_ws,
                              size_t ws_size, hipStream_t stream) {
  (void)in_sizes; (void)n_in; (void)out_size; (void)ws_size;
  const float* x    = (const float*)d_in[0];  // [B, CIN, Z]
  const float* Wqkv = (const float*)d_in[1];  // [O3, CIN]
  const float* Wout = (const float*)d_in[2];  // [512, HID]
  const float* bout = (const float*)d_in[3];  // [512]
  float* out = (float*)d_out;                 // [B, 512, Z]

  __bf16* qkv = (__bf16*)d_ws;                    // B*O3*Z bf16
  __bf16* att = qkv + (size_t)BATCH * O3 * ZLEN;  // B*HID*Z bf16

  // 1) qkv[b,o,z] = sum_c Wqkv[o,c] * x[b,c,z]  (QSCALE folded into o<512)
  dim3 g1(ZLEN / 128, O3 / 64, BATCH);
  gemm_bf16_kernel<float, /*OUT_BF16=*/true, /*BIAS=*/false, /*SCALE=*/true>
      <<<g1, 256, 0, stream>>>(Wqkv, x, qkv, nullptr, O3, ZLEN, CIN);

  // 2) softmax over z on k rows (in place)
  softmax_k_kernel<<<BATCH * HID, 256, 0, stream>>>(qkv);

  // 3) per-(b,h): ctx = k·v^T ; out = ctx^T·q
  attention_kernel<<<BATCH * HEADS, 256, 0, stream>>>(qkv, att);

  // 4) out[b,o,z] = sum_e Wout[o,e] * att[b,e,z] + bout[o]
  dim3 g2(ZLEN / 128, HID / 64, BATCH);
  gemm_bf16_kernel<__bf16, /*OUT_BF16=*/false, /*BIAS=*/true, /*SCALE=*/false>
      <<<g2, 256, 0, stream>>>(Wout, att, out, bout, HID, ZLEN, HID);
}
